// CausalSelfAttention_38869454029052
// MI455X (gfx1250) — compile-verified, matrix operand order checked by reading, not run on hardware
//
#include <hip/hip_runtime.h>
#include <hip/hip_bf16.h>
#include <math.h>

// CDNA5 bf16 WMMA types
typedef __bf16 bf16;
typedef __attribute__((ext_vector_type(16))) __bf16 v16bf;
typedef __attribute__((ext_vector_type(8)))  float  v8f;

union FragU { v16bf v; uint4 u[2]; };          // 32 bytes: one wave32 bf16 fragment per lane
union Pack4 { uint2 u; bf16 h[4]; };
union Pack8 { uint4 u; bf16 h[8]; };

#define WMMA_BF16(a, b, c) \
  __builtin_amdgcn_wmma_f32_16x16x32_bf16(false, (a), false, (b), (short)0, (c), false, false)

// Drain ASYNCcnt (async global->LDS DMA) before barriers.
static __device__ __forceinline__ void wait_async0() {
#if __has_builtin(__builtin_amdgcn_s_wait_asynccnt)
  __builtin_amdgcn_s_wait_asynccnt(0);
#else
  asm volatile("s_wait_asynccnt 0x0" ::: "memory");
#endif
}

// Async 16B global->LDS copy (GLOBAL_LOAD_ASYNC_TO_LDS_B128, ASYNCcnt).
// lds = LDS byte address (low 32 bits of flat shared address), ga = global addr.
static __device__ __forceinline__ void async_b128(unsigned lds, const void* gp) {
  unsigned long long ga = (unsigned long long)(size_t)gp;
  asm volatile("global_load_async_to_lds_b128 %0, %1, off" :: "v"(lds), "v"(ga) : "memory");
}

// ISA 16-bit fragment swizzle (16x32 tile, wave32):
//   lane = (m_or_n & 15) + 16*half,  half = (k>>3)&1
//   per-lane element index idx = (k&7) | ((k&16)>>1)   (k in 0..31)
// LDS layout: [tile][chunk(=idx>>3)][lane] of uint4  ->  fragment = 2x ds_load_b128.

// ---------------------------------------------------------------------------
// Kernel 1: QKV = X[8192,1024] @ Wqkv[1024,3072] + b ; scatter to q/k/v bf16
// [B,H,T,Dh] buffers (q pre-scaled by 1/sqrt(Dh)).
// Block: 256 thr (8 waves), tile 128x128, K-step 32. Wave = 32x64 (2x4 tiles).
// ---------------------------------------------------------------------------
__global__ __launch_bounds__(256) void qkv_gemm_kernel(
    const float* __restrict__ X, const float* __restrict__ W,
    const float* __restrict__ bias,
    bf16* __restrict__ qb, bf16* __restrict__ kb, bf16* __restrict__ vb) {
  __shared__ __align__(16) uint4 Af[8][2][32];  // 8 m-tiles of 16x32
  __shared__ __align__(16) uint4 Bf[8][2][32];  // 8 n-tiles of 32x16

  const int tid = threadIdx.x;
  const int lane = tid & 31, wave = tid >> 5;
  const int half = lane >> 4, lanelow = lane & 15;
  const int wm = wave >> 1, wn = wave & 1;
  const int bm = blockIdx.x * 128, bn = blockIdx.y * 128;

  v8f acc[2][4];
#pragma unroll
  for (int i = 0; i < 2; i++)
#pragma unroll
    for (int j = 0; j < 4; j++) { v8f z = {}; acc[i][j] = z; }

  for (int kt = 0; kt < 1024; kt += 32) {
    if (kt + 32 < 1024) {  // global_prefetch_b8 of next K-slice (WGP scope)
      __builtin_prefetch(X + (size_t)(bm + (tid >> 1)) * 1024 + kt + 32, 0, 3);
      __builtin_prefetch(W + (size_t)(kt + 32 + (tid >> 3)) * 3072 + bn + (tid & 7) * 16, 0, 3);
    }
    // Stage X tile 128x32 (fp32 -> bf16), swizzled A-fragment layout.
#pragma unroll
    for (int i = 0; i < 4; i++) {
      int f4 = tid + 256 * i;          // 1024 float4
      int row = f4 >> 3, c4 = f4 & 7;  // 8 float4 per row
      int k0 = c4 * 4;
      float4 v = *(const float4*)(X + (size_t)(bm + row) * 1024 + kt + k0);
      Pack4 pk;
      pk.h[0] = (bf16)v.x; pk.h[1] = (bf16)v.y;
      pk.h[2] = (bf16)v.z; pk.h[3] = (bf16)v.w;
      int hf = (k0 >> 3) & 1;
      int idx0 = (k0 & 7) | ((k0 & 16) >> 1);        // 0,4,8,12
      ((uint2*)&Af[row >> 4][idx0 >> 3][(row & 15) + 16 * hf])[(idx0 & 7) >> 2] = pk.u;
    }
    // Stage W tile 32x128: each thread takes 4 k-values of one column so the
    // converted quad lands in one contiguous swizzled chunk (packed b64 store).
#pragma unroll
    for (int i = 0; i < 4; i++) {
      int task = tid + 256 * i;            // 1024 tasks: 128 n x 8 k-quads
      int n = task & 127, k0 = (task >> 7) * 4;
      const float* wp = W + (size_t)(kt + k0) * 3072 + bn + n;
      Pack4 pk;
      pk.h[0] = (bf16)wp[0];
      pk.h[1] = (bf16)wp[3072];
      pk.h[2] = (bf16)wp[2 * 3072];
      pk.h[3] = (bf16)wp[3 * 3072];
      int hf = (k0 >> 3) & 1;
      int idx0 = (k0 & 7) | ((k0 & 16) >> 1);
      ((uint2*)&Bf[n >> 4][idx0 >> 3][(n & 15) + 16 * hf])[(idx0 & 7) >> 2] = pk.u;
    }
    __syncthreads();

    FragU afr[2];
#pragma unroll
    for (int im = 0; im < 2; im++) {
      afr[im].u[0] = Af[wm * 2 + im][0][lane];
      afr[im].u[1] = Af[wm * 2 + im][1][lane];
    }
#pragma unroll
    for (int jn = 0; jn < 4; jn++) {
      FragU bfr;
      bfr.u[0] = Bf[wn * 4 + jn][0][lane];
      bfr.u[1] = Bf[wn * 4 + jn][1][lane];
#pragma unroll
      for (int im = 0; im < 2; im++)
        acc[im][jn] = WMMA_BF16(afr[im].v, bfr.v, acc[im][jn]);
    }
    __syncthreads();
  }

  // Epilogue: bias, scatter into q/k/v [B,H,T,Dh]; fold softmax scale into q.
#pragma unroll
  for (int im = 0; im < 2; im++)
#pragma unroll
    for (int jn = 0; jn < 4; jn++) {
      int n = bn + wn * 64 + jn * 16 + lanelow;  // 0..3071
      float bv = bias[n];
      int which = n >> 10, c = n & 1023;
      int head = c >> 6, d = c & 63;
      bf16* dst = (which == 0) ? qb : (which == 1) ? kb : vb;
      float scl = (which == 0) ? 0.125f : 1.0f;  // 1/sqrt(64)
#pragma unroll
      for (int r = 0; r < 8; r++) {
        int m = bm + wm * 32 + im * 16 + r + 8 * half;  // 0..8191
        int bi = m >> 11, t = m & 2047;
        float val = (acc[im][jn][r] + bv) * scl;
        dst[((size_t)((bi * 16 + head) * 2048 + t) << 6) + d] = (bf16)val;
      }
    }
}

// ---------------------------------------------------------------------------
// Kernel 2: causal flash attention. Block = (b,h) x 64-query tile, 4 waves.
// Wave owns 16 query rows; Q kept as WMMA A-fragments in registers.
// Q/K tiles staged with async global->LDS DMA directly into fragment layout.
// ---------------------------------------------------------------------------
__global__ __launch_bounds__(128) void attn_kernel(
    const bf16* __restrict__ qb, const bf16* __restrict__ kb,
    const bf16* __restrict__ vb, bf16* __restrict__ yb) {
  __shared__ __align__(16) uint4 Qf[4][2][2][32];  // [mtile][kk][chunk][lane]
  __shared__ __align__(16) uint4 Kf[4][2][2][32];  // [key-tile][kk(dh)][chunk][lane]
  __shared__ __align__(16) bf16  Vt[64][72];       // transposed V: [dh][key]
  __shared__ __align__(16) uint4 Pf[4][2][2][32];  // [wave][kk(key)][chunk][lane]

  const int tid = threadIdx.x;
  const int lane = tid & 31, wave = tid >> 5;  // wave 0..3
  const int half = lane >> 4, lanelow = lane & 15;
  const int bh = blockIdx.x;  // 0..63 (B*H)
  const int bq = blockIdx.y;  // 0..31 query tile
  const size_t base = (size_t)bh * 2048 * 64;

  // Stage Q tile (64x64 bf16) into A-fragment layout via async DMA.
#pragma unroll
  for (int i = 0; i < 4; i++) {
    int id = tid + 128 * i;          // 512 uint4
    int row = id >> 3, c8 = id & 7;  // 8 uint4 per row
    int d0 = c8 * 8;
    unsigned lds = (unsigned)(size_t)
        &Qf[row >> 4][d0 >> 5][(d0 & 16) >> 4][(row & 15) + 16 * ((d0 >> 3) & 1)];
    async_b128(lds, qb + base + (size_t)(bq * 64 + row) * 64 + d0);
  }
  wait_async0();
  __syncthreads();

  FragU qf[2];
#pragma unroll
  for (int kk = 0; kk < 2; kk++) {
    qf[kk].u[0] = Qf[wave][kk][0][lane];
    qf[kk].u[1] = Qf[wave][kk][1][lane];
  }

  v8f o[4];
#pragma unroll
  for (int ns = 0; ns < 4; ns++) { v8f z = {}; o[ns] = z; }
  float m_i[8], l_i[8];
#pragma unroll
  for (int r = 0; r < 8; r++) { m_i[r] = -1e30f; l_i[r] = 0.0f; }

  for (int jt = 0; jt <= bq; jt++) {
    __syncthreads();  // previous tile fully consumed
    if (jt < bq) {    // prefetch next K/V tile into near caches
      __builtin_prefetch(kb + base + (size_t)(jt * 64 + 64 + (tid >> 1)) * 64, 0, 3);
      __builtin_prefetch(vb + base + (size_t)(jt * 64 + 64 + (tid >> 1)) * 64, 0, 3);
    }
#pragma unroll
    for (int i = 0; i < 4; i++) {
      int id = tid + 128 * i;
      int row = id >> 3, c8 = id & 7;  // row = key 0..63
      int d0 = c8 * 8;
      // K: async DMA straight into swizzled B-fragment layout.
      unsigned lds = (unsigned)(size_t)
          &Kf[row >> 4][d0 >> 5][(d0 & 16) >> 4][(row & 15) + 16 * ((d0 >> 3) & 1)];
      async_b128(lds, kb + base + (size_t)(jt * 64 + row) * 64 + d0);
      // V: register transpose into Vt[dh][key].
      Pack8 uv;
      uv.u = *(const uint4*)(vb + base + (size_t)(jt * 64 + row) * 64 + d0);
#pragma unroll
      for (int j = 0; j < 8; j++) Vt[d0 + j][row] = uv.h[j];
    }
    wait_async0();
    __syncthreads();

    // S = Q @ K^T  (16 x 64 per wave); B-fragment straight from Kf.
    v8f s[4];
#pragma unroll
    for (int ns = 0; ns < 4; ns++) { v8f z = {}; s[ns] = z; }
#pragma unroll
    for (int ns = 0; ns < 4; ns++)
#pragma unroll
      for (int kk = 0; kk < 2; kk++) {
        FragU bfr;
        bfr.u[0] = Kf[ns][kk][0][lane];
        bfr.u[1] = Kf[ns][kk][1][lane];
        s[ns] = WMMA_BF16(qf[kk].v, bfr.v, s[ns]);
      }

    // Causal mask on diagonal tile
    if (jt == bq) {
#pragma unroll
      for (int ns = 0; ns < 4; ns++)
#pragma unroll
        for (int r = 0; r < 8; r++) {
          int qi = wave * 16 + r + 8 * half;
          int kj = ns * 16 + lanelow;
          if (kj > qi) s[ns][r] = -1e30f;
        }
    }

    // Online softmax. C-layout row lives across a 16-lane half-group.
#pragma unroll
    for (int r = 0; r < 8; r++) {
      float mx = s[0][r];
#pragma unroll
      for (int ns = 1; ns < 4; ns++) mx = fmaxf(mx, s[ns][r]);
#pragma unroll
      for (int off = 8; off >= 1; off >>= 1)
        mx = fmaxf(mx, __shfl_xor(mx, off, 32));
      float mnew = fmaxf(m_i[r], mx);
      float corr = __expf(m_i[r] - mnew);
      m_i[r] = mnew;
      float rs = 0.0f;
#pragma unroll
      for (int ns = 0; ns < 4; ns++) {
        float p = __expf(s[ns][r] - mnew);
        s[ns][r] = p;
        rs += p;
      }
#pragma unroll
      for (int off = 8; off >= 1; off >>= 1)
        rs += __shfl_xor(rs, off, 32);
      l_i[r] = l_i[r] * corr + rs;
#pragma unroll
      for (int ns = 0; ns < 4; ns++) o[ns][r] *= corr;
    }

    // Stage P (C-layout -> swizzled A-fragment layout). Scalar stores are
    // unavoidable (row elements live across lanes), but the re-read is 2xb128.
    bf16* pflat = (bf16*)&Pf[wave][0][0][0];
#pragma unroll
    for (int ns = 0; ns < 4; ns++)
#pragma unroll
      for (int r = 0; r < 8; r++) {
        int kk = ns >> 1;                       // key>>5
        int chunk = ns & 1;                     // (key'&16)>>4
        int within = lanelow & 7;               // key'&7
        int lane2 = (r + 8 * half) + 16 * (lanelow >> 3);
        pflat[((kk * 2 + chunk) * 32 + lane2) * 8 + within] = (bf16)s[ns][r];
      }

    // O += P @ V ; V B-fragment read transposed from Vt.
    FragU pf[2];
#pragma unroll
    for (int kk = 0; kk < 2; kk++) {
      pf[kk].u[0] = Pf[wave][kk][0][lane];
      pf[kk].u[1] = Pf[wave][kk][1][lane];
    }
#pragma unroll
    for (int ns = 0; ns < 4; ns++)
#pragma unroll
      for (int kk = 0; kk < 2; kk++) {
        FragU bfr;
        const bf16* vrow = &Vt[ns * 16 + lanelow][0];
        bfr.u[0] = *(const uint4*)(vrow + 32 * kk + 8 * half);
        bfr.u[1] = *(const uint4*)(vrow + 32 * kk + 8 * half + 16);
        o[ns] = WMMA_BF16(pf[kk].v, bfr.v, o[ns]);
      }
  }

  // Normalize and write y in [B*T, C] bf16 layout (heads re-merged).
  const int b = bh >> 4, h = bh & 15;
#pragma unroll
  for (int ns = 0; ns < 4; ns++)
#pragma unroll
    for (int r = 0; r < 8; r++) {
      int t = bq * 64 + wave * 16 + r + 8 * half;
      int col = h * 64 + ns * 16 + lanelow;
      float val = o[ns][r] / l_i[r];
      yb[(size_t)(b * 2048 + t) * 1024 + col] = (bf16)val;
    }
}

// ---------------------------------------------------------------------------
// Kernel 3: out = Y[8192,1024](bf16) @ Wproj[1024,1024] + b  -> fp32 d_out
// Y tile staged with async global->LDS DMA (already bf16, no conversion).
// ---------------------------------------------------------------------------
__global__ __launch_bounds__(256) void proj_gemm_kernel(
    const bf16* __restrict__ Y, const float* __restrict__ W,
    const float* __restrict__ bias, float* __restrict__ out) {
  __shared__ __align__(16) uint4 Af[8][2][32];
  __shared__ __align__(16) uint4 Bf[8][2][32];

  const int tid = threadIdx.x;
  const int lane = tid & 31, wave = tid >> 5;
  const int half = lane >> 4, lanelow = lane & 15;
  const int wm = wave >> 1, wn = wave & 1;
  const int bm = blockIdx.x * 128, bn = blockIdx.y * 128;

  v8f acc[2][4];
#pragma unroll
  for (int i = 0; i < 2; i++)
#pragma unroll
    for (int j = 0; j < 4; j++) { v8f z = {}; acc[i][j] = z; }

  for (int kt = 0; kt < 1024; kt += 32) {
    if (kt + 32 < 1024) {
      __builtin_prefetch(Y + (size_t)(bm + (tid >> 1)) * 1024 + kt + 32, 0, 3);
      __builtin_prefetch(W + (size_t)(kt + 32 + (tid >> 3)) * 1024 + bn + (tid & 7) * 16, 0, 3);
    }
    // Stage Y tile 128x32 via async DMA into swizzled A-fragment layout.
#pragma unroll
    for (int i = 0; i < 2; i++) {
      int id = tid + 256 * i;          // 512 uint4
      int row = id >> 2, c8 = id & 3;  // 4 uint4 per row
      int k0 = c8 * 8;
      unsigned lds = (unsigned)(size_t)
          &Af[row >> 4][(k0 & 16) >> 4][(row & 15) + 16 * ((k0 >> 3) & 1)];
      async_b128(lds, Y + (size_t)(bm + row) * 1024 + kt + k0);
    }
    // Stage W tile 32x128 (fp32 -> bf16), k-quad per thread, packed store.
#pragma unroll
    for (int i = 0; i < 4; i++) {
      int task = tid + 256 * i;
      int n = task & 127, k0 = (task >> 7) * 4;
      const float* wp = W + (size_t)(kt + k0) * 1024 + bn + n;
      Pack4 pk;
      pk.h[0] = (bf16)wp[0];
      pk.h[1] = (bf16)wp[1024];
      pk.h[2] = (bf16)wp[2 * 1024];
      pk.h[3] = (bf16)wp[3 * 1024];
      int hf = (k0 >> 3) & 1;
      int idx0 = (k0 & 7) | ((k0 & 16) >> 1);
      ((uint2*)&Bf[n >> 4][idx0 >> 3][(n & 15) + 16 * hf])[(idx0 & 7) >> 2] = pk.u;
    }
    wait_async0();
    __syncthreads();

    FragU afr[2];
#pragma unroll
    for (int im = 0; im < 2; im++) {
      afr[im].u[0] = Af[wm * 2 + im][0][lane];
      afr[im].u[1] = Af[wm * 2 + im][1][lane];
    }
#pragma unroll
    for (int jn = 0; jn < 4; jn++) {
      FragU bfr;
      bfr.u[0] = Bf[wn * 4 + jn][0][lane];
      bfr.u[1] = Bf[wn * 4 + jn][1][lane];
#pragma unroll
      for (int im = 0; im < 2; im++)
        acc[im][jn] = WMMA_BF16(afr[im].v, bfr.v, acc[im][jn]);
    }
    __syncthreads();
  }

#pragma unroll
  for (int im = 0; im < 2; im++)
#pragma unroll
    for (int jn = 0; jn < 4; jn++) {
      int n = bn + wn * 64 + jn * 16 + lanelow;
      float bv = bias[n];
#pragma unroll
      for (int r = 0; r < 8; r++) {
        int m = bm + wm * 32 + im * 16 + r + 8 * half;
        out[(size_t)m * 1024 + n] = acc[im][jn][r] + bv;
      }
    }
}

// ---------------------------------------------------------------------------
extern "C" void kernel_launch(void* const* d_in, const int* in_sizes, int n_in,
                              void* d_out, int out_size, void* d_ws, size_t ws_size,
                              hipStream_t stream) {
  (void)in_sizes; (void)n_in; (void)out_size; (void)ws_size;
  const float* x      = (const float*)d_in[0];
  const float* w_qkv  = (const float*)d_in[1];
  const float* b_qkv  = (const float*)d_in[2];
  const float* w_proj = (const float*)d_in[3];
  const float* b_proj = (const float*)d_in[4];
  float* out = (float*)d_out;

  const size_t elems = (size_t)4 * 16 * 2048 * 64;  // 8,388,608 per buffer
  bf16* qb = (bf16*)d_ws;
  bf16* kb = qb + elems;
  bf16* vb = kb + elems;
  bf16* yb = vb + elems;  // total 67.1 MB workspace

  dim3 g1(64, 24);
  qkv_gemm_kernel<<<g1, dim3(256), 0, stream>>>(x, w_qkv, b_qkv, qb, kb, vb);
  dim3 g2(64, 32);
  attn_kernel<<<g2, dim3(128), 0, stream>>>(qb, kb, vb, yb);
  dim3 g3(64, 8);
  proj_gemm_kernel<<<g3, dim3(256), 0, stream>>>(yb, w_proj, b_proj, out);
}
